// RecurrentLegendreLayer_1743756722949
// MI455X (gfx1250) — compile-verified, matrix operand order checked by reading, not run on hardware
//
#include <hip/hip_runtime.h>

typedef __attribute__((ext_vector_type(16))) __bf16 v16bf;
typedef __attribute__((ext_vector_type(8)))  float  v8f;
typedef __attribute__((ext_vector_type(4)))  int    v4i;

#define MAXDEG 8
#define NF (MAXDEG + 1)
#define B_ROWS 16384
#define I_DIM 512
#define O_DIM 512

#define BM 128
#define BN 128
#define KC 32            // i-chunk (K per WMMA group)
#define LDA 40           // padded LDS row stride (elems): 80B rows, 16B aligned
#define LDB 40

// prepass: weights f32 -> bf16 via native hardware convert (RNE)
__global__ void cvt_weights_bf16(const float* __restrict__ w,
                                 __bf16* __restrict__ wbf, int n) {
  int i = blockIdx.x * blockDim.x + threadIdx.x;
  int stride = gridDim.x * blockDim.x;
  for (; i < n; i += stride) wbf[i] = (__bf16)w[i];
}

__global__ __launch_bounds__(256) void legendre_wmma_gemm(
    const float* __restrict__ x, const __bf16* __restrict__ wbf,
    const float* __restrict__ bias, float* __restrict__ out) {
  // double-buffered tiles: one barrier per K-step
  __shared__ __bf16         A_lds[2][BM * LDA];   // [row][k]  polys
  __shared__ unsigned short B_lds[2][BN * LDB];   // [col][k]  weights (bits)

  const int tid  = threadIdx.x;
  const int lane = tid & 31;
  const int wave = tid >> 5;         // 8 waves
  const int wm   = wave >> 1;        // 0..3  (M sub-block of 32 rows)
  const int wn   = wave & 1;         // 0..1  (N sub-block of 64 cols)
  const int m_blk = blockIdx.x * BM;
  const int n_blk = blockIdx.y * BN;

  // poly-producer mapping: each thread owns 16 contiguous i's of one row
  const int prow = tid >> 1;             // 0..127
  const int pi0  = (tid & 1) * 16;       // 0 or 16
  // B-loader mapping: lane = K-row (wide global load), wave = 16-col group
  const int bk  = tid & 31;              // 0..31  k within chunk
  const int bn8 = (tid >> 5) * 16;       // 0,16,...,112 col group

  const int khi = ((lane >> 4) & 1) * 8; // hi half-wave: K offset +8
  const int am  = lane & 15;
  const int bnl = lane & 15;

  // Legendre: P_f = C1[f]*x*P_{f-1} - C2[f]*P_{f-2}; literals after unroll
  constexpr float C1[NF] = {0.f, 0.f, 3.f / 2.f, 5.f / 3.f, 7.f / 4.f,
                            9.f / 5.f, 11.f / 6.f, 13.f / 7.f, 15.f / 8.f};
  constexpr float C2[NF] = {0.f, 0.f, 1.f / 2.f, 2.f / 3.f, 3.f / 4.f,
                            4.f / 5.f, 5.f / 6.f, 6.f / 7.f, 7.f / 8.f};

  v8f acc[2][4];
  const v8f vzero = {0.f, 0.f, 0.f, 0.f, 0.f, 0.f, 0.f, 0.f};
#pragma unroll
  for (int mi = 0; mi < 2; ++mi)
#pragma unroll
    for (int ni = 0; ni < 4; ++ni) acc[mi][ni] = vzero;

  for (int ic = 0; ic < I_DIM / KC; ++ic) {
    float xv[16], p1[16], p2[16];
    const float* xp = x + (size_t)(m_blk + prow) * I_DIM + ic * KC + pi0;
#pragma unroll
    for (int j = 0; j < 16; ++j) xv[j] = xp[j];

#pragma unroll
    for (int f = 0; f < NF; ++f) {
      const int p = (ic + f) & 1;  // buffer parity of step ic*9+f (9 is odd)

      // advance Legendre recurrence in registers (f32); branches fold away
      float cur[16];
      if (f == 0) {
#pragma unroll
        for (int j = 0; j < 16; ++j) { cur[j] = 1.0f; p2[j] = 1.0f; }
      } else if (f == 1) {
#pragma unroll
        for (int j = 0; j < 16; ++j) { cur[j] = xv[j]; p1[j] = xv[j]; }
      } else {
        const float c1 = C1[f];
        const float c2 = C2[f];
#pragma unroll
        for (int j = 0; j < 16; ++j) {
          float pn = c1 * xv[j] * p1[j] - c2 * p2[j];
          cur[j] = pn;
          p2[j] = p1[j];
          p1[j] = pn;
        }
      }

      // stage A tile [BM][KC]: native f32->bf16 converts (packed RNE)
      {
        __bf16* arow = &A_lds[p][prow * LDA + pi0];
#pragma unroll
        for (int j = 0; j < 16; ++j) arow[j] = (__bf16)cur[j];
      }
      // stage B tile K-transposed: 2 x b128 global loads, b16 scatter to LDS.
      // B_lds[p][n][k] = W'[(f*512 + ic*32 + k)][n_blk + n]
      {
        const size_t grow = (size_t)(f * I_DIM + ic * KC + bk) * O_DIM + n_blk + bn8;
        union { v4i v; unsigned short s[8]; } c0, c1;
        const v4i* gp = (const v4i*)(wbf + grow);   // 32B-aligned
        c0.v = gp[0];
        c1.v = gp[1];
        unsigned short* bb = &B_lds[p][0];
#pragma unroll
        for (int j = 0; j < 8; ++j) bb[(bn8 + j) * LDB + bk] = c0.s[j];
#pragma unroll
        for (int j = 0; j < 8; ++j) bb[(bn8 + 8 + j) * LDB + bk] = c1.s[j];
      }

      __syncthreads();  // tile p ready; other buffer still owned by prev step

      // build fragments per ISA 16-bit 16x32 layout:
      // VGPR v holds K = (v<4 ? 2v : 2v+8) (+8 on hi half-wave), pairs (K,K+1)
      v16bf af[2];
      union { v16bf v; unsigned short u[16]; } bfr[4];
#pragma unroll
      for (int mi = 0; mi < 2; ++mi) {
        const int row = wm * 32 + mi * 16 + am;
#pragma unroll
        for (int v = 0; v < 8; ++v) {
          const int k0 = ((v & 4) ? (2 * v + 8) : (2 * v)) + khi;
          af[mi][2 * v]     = A_lds[p][row * LDA + k0];
          af[mi][2 * v + 1] = A_lds[p][row * LDA + k0 + 1];
        }
      }
#pragma unroll
      for (int ni = 0; ni < 4; ++ni) {
        const int col = wn * 64 + ni * 16 + bnl;
#pragma unroll
        for (int v = 0; v < 8; ++v) {
          const int k0 = ((v & 4) ? (2 * v + 8) : (2 * v)) + khi;
          bfr[ni].u[2 * v]     = B_lds[p][col * LDB + k0];
          bfr[ni].u[2 * v + 1] = B_lds[p][col * LDB + k0 + 1];
        }
      }

#pragma unroll
      for (int mi = 0; mi < 2; ++mi)
#pragma unroll
        for (int ni = 0; ni < 4; ++ni)
          acc[mi][ni] = __builtin_amdgcn_wmma_f32_16x16x32_bf16(
              false, af[mi], false, bfr[ni].v, (short)0, acc[mi][ni],
              false, false);
    }
  }

  // epilogue: C layout — VGPR r: rows r / r+8, lanes 0-15 / 16-31 share N
  const int rhi = (lane >> 4) * 8;
#pragma unroll
  for (int mi = 0; mi < 2; ++mi) {
#pragma unroll
    for (int ni = 0; ni < 4; ++ni) {
      const int col = n_blk + wn * 64 + ni * 16 + (lane & 15);
      const float bv = bias[col];
#pragma unroll
      for (int r = 0; r < 8; ++r) {
        const int row = m_blk + wm * 32 + mi * 16 + rhi + r;
        out[(size_t)row * O_DIM + col] = acc[mi][ni][r] + bv;
      }
    }
  }
}

extern "C" void kernel_launch(void* const* d_in, const int* in_sizes, int n_in,
                              void* d_out, int out_size, void* d_ws,
                              size_t ws_size, hipStream_t stream) {
  const float* x    = (const float*)d_in[0];
  const float* w    = (const float*)d_in[1];
  const float* bias = (const float*)d_in[2];
  float* out = (float*)d_out;

  __bf16* wbf = (__bf16*)d_ws;  // 9*512*512*2 = 4.7 MB
  const int nw = NF * I_DIM * O_DIM;

  cvt_weights_bf16<<<2304, 256, 0, stream>>>(w, wbf, nw);

  dim3 grid(B_ROWS / BM, O_DIM / BN);  // 128 x 4
  legendre_wmma_gemm<<<grid, 256, 0, stream>>>(x, wbf, bias, out);
}
